// Attention_50268297232732
// MI455X (gfx1250) — matrix operand order repaired and checked by reading, not compiled, not run on hardware
//
#include <hip/hip_runtime.h>

// ---------------------------------------------------------------------------
// Re-attention block for MI455X (gfx1250, wave32, WMMA).
// - All GEMMs: v_wmma_f32_16x16x32_bf16 (bf16 in, fp32 accumulate)
// - Tile staging: global_load_async_to_lds_b128 + ASYNCcnt double buffering
// - LDS rows padded to kill bank conflicts on fragment b128 reads
// ---------------------------------------------------------------------------

typedef unsigned short u16;
typedef unsigned long long u64;
typedef __attribute__((ext_vector_type(16))) __bf16 v16bf;
typedef __attribute__((ext_vector_type(8)))  float  v8f;

union Frag16 {            // one WMMA bf16 operand fragment (32B / lane)
    v16bf bf;
    uint4 q[2];
};

__device__ __forceinline__ u16 f2bf(float f) {
    union { float f; unsigned u; } c; c.f = f;
    unsigned u = c.u;
    unsigned r = u + 0x7FFFu + ((u >> 16) & 1u);   // round-to-nearest-even
    return (u16)(r >> 16);
}

// CDNA5 async global->LDS copy (ASYNCcnt-tracked, no VGPR round trip)
__device__ __forceinline__ void async_b128(unsigned lds_byte_off, u64 gaddr) {
    asm volatile("global_load_async_to_lds_b128 %0, %1, off"
                 :: "v"(lds_byte_off), "v"(gaddr) : "memory");
}

// Problem dims
constexpr int Bb    = 8;
constexpr int Nn    = 1024;
constexpr int DIM   = 768;
constexpr int Hh    = 12;
constexpr int HD    = 64;
constexpr int INNER = Hh * HD;       // 768
constexpr int K3    = 3 * INNER;     // 2304
constexpr int Mrows = Bb * Nn;       // 8192
// fold softmax scale AND log2(e) into q so softmax can use exp2
constexpr float QSCALE = 0.125f * 1.44269504088896340736f; // HD^-0.5 * log2e

// padded LDS row strides (in u16 elements) -> conflict-free b128 reads
constexpr int GSTR = 40;   // GEMM tiles: 32 data + 8 pad   (80B rows)
constexpr int KSTR = 72;   // attn K tile: 64 data + 8 pad  (144B rows)
constexpr int VSTR = 40;   // attn V tile: 32 data + 8 pad  (80B rows)
constexpr int PSTR = 40;   // attn P scratch rows

// ---------------------------------------------------------------------------
// fp32 -> bf16, 8 elements / thread, vectorized
__global__ void k_conv8(const float* __restrict__ src, u16* __restrict__ dst, int n8) {
    int i = blockIdx.x * blockDim.x + threadIdx.x;
    if (i < n8) {
        const float4* s = (const float4*)src + (size_t)i * 2;
        float4 a = s[0], b = s[1];
        union { u16 h[8]; uint4 q; } o;
        o.h[0] = f2bf(a.x); o.h[1] = f2bf(a.y); o.h[2] = f2bf(a.z); o.h[3] = f2bf(a.w);
        o.h[4] = f2bf(b.x); o.h[5] = f2bf(b.y); o.h[6] = f2bf(b.z); o.h[7] = f2bf(b.w);
        ((uint4*)dst)[i] = o.q;
    }
}

// dst[n*K + k] = bf16(src[k*N + n]) via 32x32 LDS tile (coalesced both sides)
__global__ void k_transpose_conv(const float* __restrict__ src, u16* __restrict__ dst,
                                 int K, int N) {
    __shared__ u16 tile[32][33];
    const int nb = blockIdx.x * 32, kb = blockIdx.y * 32;
    const int tx = threadIdx.x & 31, ty = threadIdx.x >> 5;   // 32 x 8
    for (int r = ty; r < 32; r += 8)
        tile[r][tx] = f2bf(src[(size_t)(kb + r) * N + nb + tx]);
    __syncthreads();
    for (int r = ty; r < 32; r += 8)
        dst[(size_t)(nb + r) * K + kb + tx] = tile[tx][r];
}

// head_scale[h] = sum over 64x64 reattn_weights[h]
__global__ void k_head_scale(const float* __restrict__ rw, float* __restrict__ hs) {
    __shared__ float red[256];
    int h = blockIdx.x, t = threadIdx.x;
    float s = 0.f;
    for (int i = t; i < HD * HD; i += 256) s += rw[h * HD * HD + i];
    red[t] = s; __syncthreads();
    for (int o = 128; o > 0; o >>= 1) {
        if (t < o) red[t] += red[t + o];
        __syncthreads();
    }
    if (t == 0) hs[h] = red[0];
}

// ---------------------------------------------------------------------------
// QKV GEMM: C[8192 x 2304] = xb @ wqkvT^T + b_qkv, epilogue scatters q/k/vT.
// WG tile 128x128, 8 waves (2x4), K-step 32, async double-buffered LDS.
__global__ __launch_bounds__(256) void k_gemm_qkv(const u16* __restrict__ xb,
                                                  const u16* __restrict__ wT,
                                                  const float* __restrict__ bias,
                                                  u16* __restrict__ qb,
                                                  u16* __restrict__ kb,
                                                  u16* __restrict__ vT) {
    __shared__ u16 As[2][128 * GSTR];
    __shared__ u16 Bs[2][128 * GSTR];
    constexpr unsigned BUFB = 128 * GSTR * 2;      // 10240 bytes per buffer
    const int n0 = blockIdx.x * 128;
    const int m0 = blockIdx.y * 128;
    const int tid  = threadIdx.x;
    const int lane = tid & 31;
    const int half = lane >> 4, l16 = lane & 15;
    const int w  = tid >> 5;
    const int mw = w & 1, nw = w >> 1;             // 2 waves in M, 4 in N
    const int lrow = tid >> 1, lseg = tid & 1;     // 32B per thread per tile

    const u16* gA = xb + (size_t)(m0 + lrow) * DIM + lseg * 16;
    const u16* gB = wT + (size_t)(n0 + lrow) * DIM + lseg * 16;
    const unsigned lA = (unsigned)(size_t)&As[0][lrow * GSTR + lseg * 16];
    const unsigned lB = (unsigned)(size_t)&Bs[0][lrow * GSTR + lseg * 16];

    auto issue = [&](int ki) {
        const unsigned bo = (unsigned)(ki & 1) * BUFB;
        const u64 ga = (u64)(size_t)(gA + ki * 32);
        const u64 gb = (u64)(size_t)(gB + ki * 32);
        async_b128(lA + bo,      ga);
        async_b128(lA + bo + 16, ga + 16);
        async_b128(lB + bo,      gb);
        async_b128(lB + bo + 16, gb + 16);
    };

    v8f acc[4][2] = {};
    issue(0);
    issue(1);

    for (int ki = 0; ki < DIM / 32; ++ki) {
        if (ki < DIM / 32 - 1) asm volatile("s_wait_asynccnt 4" ::: "memory");
        else                   asm volatile("s_wait_asynccnt 0" ::: "memory");
        __syncthreads();                      // whole tile visible to all waves

        const u16* as = &As[ki & 1][0];
        const u16* bs = &Bs[ki & 1][0];

        Frag16 af[4];
#pragma unroll
        for (int rt = 0; rt < 4; ++rt) {
            const uint4* ap = (const uint4*)&as[(mw * 64 + rt * 16 + l16) * GSTR];
            af[rt].q[0] = ap[half * 2];       // K = half*16 .. +7
            af[rt].q[1] = ap[half * 2 + 1];   // K = half*16+8 .. +15
        }
        Frag16 bfv[2];
#pragma unroll
        for (int ct = 0; ct < 2; ++ct) {
            const uint4* bp = (const uint4*)&bs[(nw * 32 + ct * 16 + l16) * GSTR + half * 16];
            bfv[ct].q[0] = bp[0];
            bfv[ct].q[1] = bp[1];
        }
#pragma unroll
        for (int rt = 0; rt < 4; ++rt)
#pragma unroll
            for (int ct = 0; ct < 2; ++ct)
                acc[rt][ct] = __builtin_amdgcn_wmma_f32_16x16x32_bf16(
                    false, af[rt].bf, false, bfv[ct].bf, (short)0, acc[rt][ct], false, false);

        __syncthreads();                      // everyone done reading this buffer
        if (ki + 2 < DIM / 32) issue(ki + 2);
    }

#pragma unroll
    for (int rt = 0; rt < 4; ++rt)
#pragma unroll
        for (int ct = 0; ct < 2; ++ct) {
            const int ncol = n0 + nw * 32 + ct * 16 + l16;
            const int sec = ncol / INNER, rem = ncol % INNER;
            const int h = rem >> 6, d = rem & 63;
            const float bv = bias[ncol];
#pragma unroll
            for (int r = 0; r < 8; ++r) {
                const int m = m0 + mw * 64 + rt * 16 + r + half * 8;
                const int b = m >> 10, n = m & 1023;
                const float v = acc[rt][ct][r] + bv;
                const size_t hb = ((size_t)(b * Hh + h) * Nn + n) * HD + d;
                if (sec == 0)      qb[hb] = f2bf(v * QSCALE);
                else if (sec == 1) kb[hb] = f2bf(v);
                else vT[((size_t)(b * Hh + h) * HD + d) * Nn + n] = f2bf(v);
            }
        }
}

// ---------------------------------------------------------------------------
// Flash-style attention. One wave owns 16 query rows. K/V chunks (32 keys)
// are cooperatively async-loaded into double-buffered LDS once per WG.
__global__ __launch_bounds__(256) void k_attn(const u16* __restrict__ qb,
                                              const u16* __restrict__ kb,
                                              const u16* __restrict__ vT,
                                              const float* __restrict__ hs,
                                              u16* __restrict__ attnb) {
    __shared__ u16 Ks[2][32 * KSTR];   // 32 keys x 64 d (padded)
    __shared__ u16 Vs[2][64 * VSTR];   // 64 d x 32 keys (padded)
    __shared__ u16 Ps[8][16 * PSTR];   // per-wave P scratch
    constexpr unsigned KBUF = 32 * KSTR * 2;    // 4608 B
    constexpr unsigned VBUF = 64 * VSTR * 2;    // 5120 B

    const int tid  = threadIdx.x;
    const int w    = tid >> 5, lane = tid & 31;
    const int half = lane >> 4, l16 = lane & 15;
    const int bh   = blockIdx.x >> 3;           // 0..95
    const int qblk = blockIdx.x & 7;
    const int b = bh / Hh, h = bh % Hh;
    const int i0 = qblk * 128 + w * 16;

    const u16* qp = qb + ((size_t)bh * Nn + i0) * HD;
    const u16* kp = kb + (size_t)bh * Nn * HD;
    const u16* vp = vT + (size_t)bh * HD * Nn;

    // cooperative async staging: K -> 1 instr/thread, V -> 1 instr/thread
    const u16* gK = kp + (size_t)(tid >> 3) * HD + (tid & 7) * 8;
    const u16* gV = vp + (size_t)(tid >> 2) * Nn + (tid & 3) * 8;
    const unsigned lK = (unsigned)(size_t)&Ks[0][(tid >> 3) * KSTR + (tid & 7) * 8];
    const unsigned lV = (unsigned)(size_t)&Vs[0][(tid >> 2) * VSTR + (tid & 3) * 8];

    auto issue = [&](int ci) {
        const unsigned buf = (unsigned)(ci & 1);
        async_b128(lK + buf * KBUF, (u64)(size_t)(gK + (size_t)ci * 32 * HD));
        async_b128(lV + buf * VBUF, (u64)(size_t)(gV + ci * 32));
    };

    // Q fragments (reused across all key chunks)
    Frag16 qf[2];
#pragma unroll
    for (int dc = 0; dc < 2; ++dc) {
        const uint4* p0 = (const uint4*)(qp + (size_t)l16 * HD + dc * 32 + half * 16);
        const uint4* p1 = (const uint4*)(qp + (size_t)l16 * HD + dc * 32 + half * 16 + 8);
        qf[dc].q[0] = p0[0];
        qf[dc].q[1] = p1[0];
    }

    float mrun[8], lrun[8];
    v8f oacc[4] = {};
#pragma unroll
    for (int r = 0; r < 8; ++r) { mrun[r] = -1e30f; lrun[r] = 0.f; }

    issue(0);
    issue(1);

    for (int ci = 0; ci < Nn / 32; ++ci) {
        if (ci < Nn / 32 - 1) asm volatile("s_wait_asynccnt 2" ::: "memory");
        else                  asm volatile("s_wait_asynccnt 0" ::: "memory");
        __syncthreads();                      // K/V chunk visible to all waves

        const u16* ks = &Ks[ci & 1][0];
        const u16* vs = &Vs[ci & 1][0];

        // --- S tiles: keys [ci*32, ci*32+16) and [+16, +32) ---
        v8f s[2];
#pragma unroll
        for (int tt = 0; tt < 2; ++tt) {
            Frag16 kf0, kf1;
            const uint4* kp0 = (const uint4*)&ks[(tt * 16 + l16) * KSTR + half * 16];
            const uint4* kp1 = (const uint4*)&ks[(tt * 16 + l16) * KSTR + 32 + half * 16];
            kf0.q[0] = kp0[0]; kf0.q[1] = kp0[1];
            kf1.q[0] = kp1[0]; kf1.q[1] = kp1[1];
            v8f z = {};
            z = __builtin_amdgcn_wmma_f32_16x16x32_bf16(false, qf[0].bf, false, kf0.bf,
                                                        (short)0, z, false, false);
            z = __builtin_amdgcn_wmma_f32_16x16x32_bf16(false, qf[1].bf, false, kf1.bf,
                                                        (short)0, z, false, false);
            s[tt] = z;
        }

        // --- online softmax (log2 domain; scale folded into q) ---
#pragma unroll
        for (int r = 0; r < 8; ++r) {
            float a = s[0][r], c = s[1][r];
            float cmax = fmaxf(a, c);
#pragma unroll
            for (int msk = 1; msk < 16; msk <<= 1)
                cmax = fmaxf(cmax, __shfl_xor(cmax, msk, 32));
            const float mnew = fmaxf(mrun[r], cmax);
            const float p0 = exp2f(a - mnew);
            const float p1 = exp2f(c - mnew);
            float psum = p0 + p1;
#pragma unroll
            for (int msk = 1; msk < 16; msk <<= 1)
                psum += __shfl_xor(psum, msk, 32);
            const float f = exp2f(mrun[r] - mnew);
            lrun[r] = lrun[r] * f + psum;
            mrun[r] = mnew;
#pragma unroll
            for (int dt = 0; dt < 4; ++dt) oacc[dt][r] *= f;
            Ps[w][(r + half * 8) * PSTR + l16]      = f2bf(p0);
            Ps[w][(r + half * 8) * PSTR + 16 + l16] = f2bf(p1);
        }
        __syncthreads();                      // P re-shaped through LDS

        // --- O += P @ V ---
        Frag16 pf;
        const uint4* pp = (const uint4*)&Ps[w][l16 * PSTR];
        pf.q[0] = pp[half * 2];
        pf.q[1] = pp[half * 2 + 1];
#pragma unroll
        for (int dt = 0; dt < 4; ++dt) {
            Frag16 vf;
            const uint4* vpp = (const uint4*)&vs[(dt * 16 + l16) * VSTR + half * 16];
            vf.q[0] = vpp[0]; vf.q[1] = vpp[1];
            oacc[dt] = __builtin_amdgcn_wmma_f32_16x16x32_bf16(
                false, pf.bf, false, vf.bf, (short)0, oacc[dt], false, false);
        }
        __syncthreads();                      // done reading this K/V buffer
        if (ci + 2 < Nn / 32) issue(ci + 2);
    }

    // --- epilogue: normalize, apply per-head reattn scale, write bf16 ---
    const float hsv = hs[h];
#pragma unroll
    for (int dt = 0; dt < 4; ++dt)
#pragma unroll
        for (int r = 0; r < 8; ++r) {
            const int i = i0 + r + half * 8;
            const float v = oacc[dt][r] / lrun[r] * hsv;
            attnb[((size_t)(b * Nn + i)) * INNER + h * HD + dt * 16 + l16] = f2bf(v);
        }
}

// ---------------------------------------------------------------------------
// Output GEMM: out[8192 x 768] = attnb @ woutT^T + b_out (fp32 result)
__global__ __launch_bounds__(256) void k_gemm_out(const u16* __restrict__ ab,
                                                  const u16* __restrict__ wT,
                                                  const float* __restrict__ bias,
                                                  float* __restrict__ out) {
    __shared__ u16 As[2][128 * GSTR];
    __shared__ u16 Bs[2][128 * GSTR];
    constexpr unsigned BUFB = 128 * GSTR * 2;
    const int n0 = blockIdx.x * 128;
    const int m0 = blockIdx.y * 128;
    const int tid  = threadIdx.x;
    const int lane = tid & 31;
    const int half = lane >> 4, l16 = lane & 15;
    const int w  = tid >> 5;
    const int mw = w & 1, nw = w >> 1;
    const int lrow = tid >> 1, lseg = tid & 1;

    const u16* gA = ab + (size_t)(m0 + lrow) * INNER + lseg * 16;
    const u16* gB = wT + (size_t)(n0 + lrow) * INNER + lseg * 16;
    const unsigned lA = (unsigned)(size_t)&As[0][lrow * GSTR + lseg * 16];
    const unsigned lB = (unsigned)(size_t)&Bs[0][lrow * GSTR + lseg * 16];

    auto issue = [&](int ki) {
        const unsigned bo = (unsigned)(ki & 1) * BUFB;
        const u64 ga = (u64)(size_t)(gA + ki * 32);
        const u64 gb = (u64)(size_t)(gB + ki * 32);
        async_b128(lA + bo,      ga);
        async_b128(lA + bo + 16, ga + 16);
        async_b128(lB + bo,      gb);
        async_b128(lB + bo + 16, gb + 16);
    };

    v8f acc[4][2] = {};
    issue(0);
    issue(1);

    for (int ki = 0; ki < INNER / 32; ++ki) {
        if (ki < INNER / 32 - 1) asm volatile("s_wait_asynccnt 4" ::: "memory");
        else                     asm volatile("s_wait_asynccnt 0" ::: "memory");
        __syncthreads();

        const u16* as = &As[ki & 1][0];
        const u16* bs = &Bs[ki & 1][0];

        Frag16 af[4];
#pragma unroll
        for (int rt = 0; rt < 4; ++rt) {
            const uint4* ap = (const uint4*)&as[(mw * 64 + rt * 16 + l16) * GSTR];
            af[rt].q[0] = ap[half * 2];
            af[rt].q[1] = ap[half * 2 + 1];
        }
        Frag16 bfv[2];
#pragma unroll
        for (int ct = 0; ct < 2; ++ct) {
            const uint4* bp = (const uint4*)&bs[(nw * 32 + ct * 16 + l16) * GSTR + half * 16];
            bfv[ct].q[0] = bp[0];
            bfv[ct].q[1] = bp[1];
        }
#pragma unroll
        for (int rt = 0; rt < 4; ++rt)
#pragma unroll
            for (int ct = 0; ct < 2; ++ct)
                acc[rt][ct] = __builtin_amdgcn_wmma_f32_16x16x32_bf16(
                    false, af[rt].bf, false, bfv[ct].bf, (short)0, acc[rt][ct], false, false);

        __syncthreads();
        if (ki + 2 < INNER / 32) issue(ki + 2);
    }

#pragma unroll
    for (int rt = 0; rt < 4; ++rt)
#pragma unroll
        for (int ct = 0; ct < 2; ++ct) {
            const int ncol = n0 + nw * 32 + ct * 16 + l16;
            const float bv = bias[ncol];
#pragma unroll
            for (int r = 0; r < 8; ++r) {
                const int m = m0 + mw * 64 + rt * 16 + r + half * 8;
                out[(size_t)m * DIM + ncol] = acc[rt][ct][r] + bv;
            }
        }
}

// ---------------------------------------------------------------------------
extern "C" void kernel_launch(void* const* d_in, const int* in_sizes, int n_in,
                              void* d_out, int out_size, void* d_ws, size_t ws_size,
                              hipStream_t stream) {
    const float* x      = (const float*)d_in[0];
    const float* w_qkv  = (const float*)d_in[1];
    const float* b_qkv  = (const float*)d_in[2];
    const float* reattn = (const float*)d_in[3];
    const float* w_out  = (const float*)d_in[4];
    const float* b_out  = (const float*)d_in[5];
    float* out = (float*)d_out;

    char* ws = (char*)d_ws;
    size_t off = 0;
    auto alloc = [&](size_t bytes) -> void* {
        void* p = ws + off;
        off = (off + bytes + 255) & ~(size_t)255;
        return p;
    };

    u16*   xb    = (u16*)alloc((size_t)Mrows * DIM * 2);
    u16*   wqkvT = (u16*)alloc((size_t)K3 * DIM * 2);
    u16*   woutT = (u16*)alloc((size_t)DIM * INNER * 2);
    u16*   qb    = (u16*)alloc((size_t)Bb * Hh * Nn * HD * 2);
    u16*   kb    = (u16*)alloc((size_t)Bb * Hh * Nn * HD * 2);
    u16*   vT    = (u16*)alloc((size_t)Bb * Hh * HD * Nn * 2);
    u16*   attnb = (u16*)alloc((size_t)Mrows * INNER * 2);
    float* hsbuf = (float*)alloc(Hh * sizeof(float));
    (void)ws_size; (void)in_sizes; (void)n_in; (void)out_size;

    // Stage 0: conversions / tiled transposes / head-scale reduction
    {
        int n8 = Mrows * DIM / 8;
        k_conv8<<<(n8 + 255) / 256, 256, 0, stream>>>(x, xb, n8);
    }
    k_transpose_conv<<<dim3(K3 / 32, DIM / 32), 256, 0, stream>>>(w_qkv, wqkvT, DIM, K3);
    k_transpose_conv<<<dim3(DIM / 32, INNER / 32), 256, 0, stream>>>(w_out, woutT, INNER, DIM);
    k_head_scale<<<Hh, 256, 0, stream>>>(reattn, hsbuf);

    // Stage 1: fused QKV projection (+bias, q pre-scaled, v transposed)
    k_gemm_qkv<<<dim3(K3 / 128, Mrows / 128), 256, 0, stream>>>(
        xb, wqkvT, b_qkv, qb, kb, vT);

    // Stage 2: flash attention with per-head reattn scale
    k_attn<<<Bb * Hh * (Nn / 128), 256, 0, stream>>>(qb, kb, vT, hsbuf, attnb);

    // Stage 3: output projection (+bias), fp32 result
    k_gemm_out<<<dim3(DIM / 128, Mrows / 128), 256, 0, stream>>>(
        attnb, woutT, b_out, out);
}
